// PointTransformerBlock_56624848831034
// MI455X (gfx1250) — matrix operand order repaired and checked by reading, not compiled, not run on hardware
//
#include <hip/hip_runtime.h>
#include <hip/hip_bf16.h>

// Point Transformer block, MI455X (gfx1250, wave32, WMMA).
// B=4, N=8192, C=MID=128, NS=16, SH=8, CS=16.

#define BB   4
#define NN_  8192
#define CC   128
#define NSK  16
#define CSK  16

typedef __attribute__((ext_vector_type(16))) _Float16 v16h;
typedef __attribute__((ext_vector_type(8)))  float    v8f;

// Inverse of the 16-bit A-matrix 16x32 fragment layout:
// element (row m, kk in 0..31) lives in lane = h*16+m, slot e.
__device__ __forceinline__ void a_swz(int kk, int m, int& lane, int& e) {
  lane = ((kk >> 3) & 1) * 16 + m;
  e    = ((kk & 16) >> 1) | (kk & 7);
}

// ---------------------------------------------------------------------------
// Kernel 1: kNN (top-16 by squared distance) — lane-per-query, candidate tiles
// staged into LDS with CDNA5 async global->LDS loads.
// ---------------------------------------------------------------------------
__global__ __launch_bounds__(256) void knn_kernel(const float* __restrict__ xyz,
                                                  int* __restrict__ knn) {
  __shared__ float tile[768];   // 256 candidates * xyz
  const int b = blockIdx.y;
  const int n = blockIdx.x * 256 + threadIdx.x;
  const float* xb = xyz + (size_t)b * NN_ * 3;
  const float qx = xb[n * 3 + 0];
  const float qy = xb[n * 3 + 1];
  const float qz = xb[n * 3 + 2];

  float dl[16]; int il[16];
#pragma unroll
  for (int i = 0; i < 16; ++i) { dl[i] = 3.4e38f; il[i] = 0; }

  for (int base = 0; base < NN_; base += 256) {
    __syncthreads();   // previous tile fully consumed
    {
      unsigned long long gb = (unsigned long long)(xb + (size_t)base * 3);
#pragma unroll
      for (int r = 0; r < 3; ++r) {
        unsigned int voff = (unsigned int)((threadIdx.x + r * 256) * 4u);
        unsigned int lds  = (unsigned int)(unsigned long long)(&tile[threadIdx.x + r * 256]);
        asm volatile("global_load_async_to_lds_b32 %0, %1, %2"
                     :: "v"(lds), "v"(voff), "s"(gb) : "memory");
      }
      asm volatile("s_wait_asynccnt 0" ::: "memory");
    }
    __syncthreads();

#pragma unroll 4
    for (int j = 0; j < 256; ++j) {
      float dx = tile[j * 3 + 0] - qx;
      float dy = tile[j * 3 + 1] - qy;
      float dz = tile[j * 3 + 2] - qz;
      float d  = dx * dx + dy * dy + dz * dz;
      if (d < dl[15]) {   // rare after warm-up: single bubble insertion
        dl[15] = d; il[15] = base + j;
#pragma unroll
        for (int s = 15; s > 0; --s) {
          if (dl[s] < dl[s - 1]) {
            float td = dl[s]; dl[s] = dl[s - 1]; dl[s - 1] = td;
            int   ti = il[s]; il[s] = il[s - 1]; il[s - 1] = ti;
          }
        }
      }
    }
  }
  int* o = knn + ((size_t)b * NN_ + n) * NSK;
#pragma unroll
  for (int i = 0; i < 16; ++i) o[i] = il[i];
}

// ---------------------------------------------------------------------------
// Kernel 2: fused Q/K/V projection GEMM (32768x128 @ 128x128 + bias), f16 WMMA
// with f32 accumulation. LDS tiles are staged directly in fragment-swizzled
// layout so every fragment read is one contiguous v16h (2x ds_load_b128).
// Workgroup: 256 threads (8 waves) -> 64 rows x 128 cols; wave = 1 col-tile,
// 4 row-tiles, K=128 as 4 steps of v_wmma_f32_16x16x32_f16.
// ---------------------------------------------------------------------------
__global__ __launch_bounds__(256) void proj_kernel(
    const float* __restrict__ px,
    const float* __restrict__ Wq, const float* __restrict__ bq,
    const float* __restrict__ Wk, const float* __restrict__ bk,
    const float* __restrict__ Wv, const float* __restrict__ bv,
    float* __restrict__ qb, float* __restrict__ kb, float* __restrict__ vb) {
  __shared__ __align__(32) _Float16 xfrag[4][4][32][16];   // 16 KB, A frags
  __shared__ __align__(32) _Float16 wsfrag[8][4][32][16];  // 32 KB, B frags

  const int z = blockIdx.z;
  const float* W    = (z == 0) ? Wq : (z == 1) ? Wk : Wv;
  const float* bias = (z == 0) ? bq : (z == 1) ? bk : bv;
  float*       Y    = (z == 0) ? qb : (z == 1) ? kb : vb;

  const int tid  = threadIdx.x;
  const int row0 = blockIdx.x * 64;

  // stage W (linear coalesced read, swizzled f16 write)
  for (int eL = 0; eL < 64; ++eL) {
    int i = eL * 256 + tid;                 // i = k*128 + col
    int k = i >> 7, col = i & 127;
    wsfrag[col >> 4][k >> 5][((k >> 3) & 1) * 16 + (col & 15)][((k & 16) >> 1) | (k & 7)] =
        (_Float16)W[i];
  }
  // stage X tile (linear coalesced read, swizzled f16 write)
  for (int eL = 0; eL < 32; ++eL) {
    int i = eL * 256 + tid;                 // i = row*128 + c
    int row = i >> 7, c = i & 127;
    int lane, e; a_swz(c & 31, row & 15, lane, e);
    xfrag[row >> 4][c >> 5][lane][e] =
        (_Float16)px[(size_t)(row0 + row) * CC + c];
  }
  __syncthreads();

  const int lane = tid & 31, wave = tid >> 5;
  const int colt = wave;                    // 8 waves = 8 col tiles

  v8f acc0 = {}, acc1 = {}, acc2 = {}, acc3 = {};
#pragma unroll
  for (int ks = 0; ks < 4; ++ks) {
    v16h bf = *(const v16h*)&wsfrag[colt][ks][lane][0];
    v16h a0 = *(const v16h*)&xfrag[0][ks][lane][0];
    v16h a1 = *(const v16h*)&xfrag[1][ks][lane][0];
    v16h a2 = *(const v16h*)&xfrag[2][ks][lane][0];
    v16h a3 = *(const v16h*)&xfrag[3][ks][lane][0];
    acc0 = __builtin_amdgcn_wmma_f32_16x16x32_f16(false, a0, false, bf, (short)0, acc0, false, false);
    acc1 = __builtin_amdgcn_wmma_f32_16x16x32_f16(false, a1, false, bf, (short)0, acc1, false, false);
    acc2 = __builtin_amdgcn_wmma_f32_16x16x32_f16(false, a2, false, bf, (short)0, acc2, false, false);
    acc3 = __builtin_amdgcn_wmma_f32_16x16x32_f16(false, a3, false, bf, (short)0, acc3, false, false);
  }

  const int nn = lane & 15, mh = lane >> 4;
  const float bcol = bias[colt * 16 + nn];
#pragma unroll
  for (int i = 0; i < 8; ++i) {
    int m = mh * 8 + i;
    Y[(size_t)(row0 +  0 + m) * CC + colt * 16 + nn] = acc0[i] + bcol;
    Y[(size_t)(row0 + 16 + m) * CC + colt * 16 + nn] = acc1[i] + bcol;
    Y[(size_t)(row0 + 32 + m) * CC + colt * 16 + nn] = acc2[i] + bcol;
    Y[(size_t)(row0 + 48 + m) * CC + colt * 16 + nn] = acc3[i] + bcol;
  }
}

// ---------------------------------------------------------------------------
// Kernel 3: attention-weight MLP. Per row (b,n,ns): w = q - k + pos_enc,
// a = relu(bn(w)) written to LDS pre-swizzled as WMMA A-fragments; first
// linear (128->16) = 4 WMMAs per wave (wave = one point's 16 neighbors),
// second linear (16->16) = one K-zero-padded WMMA whose A-fragment is
// scattered directly from the first WMMA's D-fragment.
// Stores pre-softmax logits (B,N,NS,CS).
// ---------------------------------------------------------------------------
__global__ __launch_bounds__(256) void attnw_kernel(
    const float* __restrict__ xyz, const int* __restrict__ knn,
    const float* __restrict__ qb, const float* __restrict__ kb,
    const float* __restrict__ pw1, const float* __restrict__ pb1,
    const float* __restrict__ pg1, const float* __restrict__ pbe1,
    const float* __restrict__ pw2, const float* __restrict__ pb2,
    const float* __restrict__ wg0, const float* __restrict__ wbe0,
    const float* __restrict__ ww1, const float* __restrict__ wb1,
    const float* __restrict__ wg1, const float* __restrict__ wbe1,
    const float* __restrict__ ww2, const float* __restrict__ wb2,
    float* __restrict__ logits) {
  __shared__ __align__(32) _Float16 wfrag[8][4][32][16];  // 32 KB, A frags (8 points)
  __shared__ __align__(32) _Float16 w1frag[4][32][16];    // 4 KB, B frags (w_w1)
  __shared__ __align__(32) _Float16 w2frag[32][16];       // 1 KB, B frag (w_w2, K-pad)
  __shared__ __align__(32) _Float16 h1frag[8][32][16];    // 8 KB, A frags (stage 2)

  const int tid = threadIdx.x;
  const int b   = blockIdx.y;
  const int p0  = blockIdx.x * 8;       // 8 points * 16 neighbors = 128 rows
  const float inv = rsqrtf(1.0f + 1e-5f);

  // stage w_w1 (2048 vals): linear read, B-swizzled write
  for (int eL = 0; eL < 8; ++eL) {
    int i = eL * 256 + tid;             // i = k*16 + col
    int k = i >> 4, col = i & 15;
    w1frag[k >> 5][((k >> 3) & 1) * 16 + col][((k & 16) >> 1) | (k & 7)] = (_Float16)ww1[i];
  }
  // stage w_w2 into B-frag layout with K padded 16->32 (slots 8..15 zero)
  for (int eL = 0; eL < 2; ++eL) {
    int i = eL * 256 + tid;             // 512 slots: [lane][e]
    int ln = i >> 4, e = i & 15;
    int k = ((e & 8) << 1) | ((ln >> 4) << 3) | (e & 7), col = ln & 15;
    w2frag[ln][e] = (k < 16) ? (_Float16)ww2[k * 16 + col] : (_Float16)0.0f;
  }

  // ---- phase 1: a = relu(bn(q - k + pos_enc)) -> A-swizzled f16 in LDS ----
  {
    const int r = tid >> 1, cpart = (tid & 1) * 64;
    const int p = r >> 4, ns = r & 15;
    const int n = p0 + p;
    const int idx = knn[((size_t)b * NN_ + n) * NSK + ns];
    const float* xb = xyz + (size_t)b * NN_ * 3;
    const float dx = xb[idx * 3 + 0] - xb[n * 3 + 0];
    const float dy = xb[idx * 3 + 1] - xb[n * 3 + 1];
    const float dz = xb[idx * 3 + 2] - xb[n * 3 + 2];
    float p1[3];
#pragma unroll
    for (int j = 0; j < 3; ++j) {
      float t = dx * pw1[0 * 3 + j] + dy * pw1[1 * 3 + j] + dz * pw1[2 * 3 + j] + pb1[j];
      p1[j] = fmaxf(t * (pg1[j] * inv) + pbe1[j], 0.0f);
    }
    const float* qrow = qb + ((size_t)b * NN_ + n)   * CC;
    const float* krow = kb + ((size_t)b * NN_ + idx) * CC;
    for (int c = 0; c < 64; ++c) {
      int cc = cpart + c;
      float pe = p1[0] * pw2[cc] + p1[1] * pw2[CC + cc] + p1[2] * pw2[2 * CC + cc] + pb2[cc];
      float wv = qrow[cc] - krow[cc] + pe;
      float a  = fmaxf(wv * (wg0[cc] * inv) + wbe0[cc], 0.0f);
      int lane, e; a_swz(cc & 31, ns, lane, e);
      wfrag[p][cc >> 5][lane][e] = (_Float16)a;
    }
  }
  __syncthreads();

  // ---- phase 2: wave w handles point p0+w: h1 = a @ w_w1 + b1 ----
  const int lane = tid & 31, wave = tid >> 5;
  v8f acc = {};
#pragma unroll
  for (int ks = 0; ks < 4; ++ks) {
    v16h bf = *(const v16h*)&w1frag[ks][lane][0];
    v16h af = *(const v16h*)&wfrag[wave][ks][lane][0];
    acc = __builtin_amdgcn_wmma_f32_16x16x32_f16(false, af, false, bf, (short)0, acc, false, false);
  }

  const int nn = lane & 15, mh = lane >> 4;
  {
    // zero the K>=16 pad slots of this lane's stage-2 A fragment
#pragma unroll
    for (int e = 8; e < 16; ++e) h1frag[wave][lane][e] = (_Float16)0.0f;
    // scatter relu(bn(h1 + b1)) from D-frag into stage-2 A-frag layout
    const float s1 = wg1[nn] * inv, be1v = wbe1[nn], b1v = wb1[nn];
#pragma unroll
    for (int i = 0; i < 8; ++i) {
      int m = mh * 8 + i;
      int tl, te; a_swz(nn, m, tl, te);    // nn < 16 -> te in 0..7
      float h1 = acc[i] + b1v;
      h1frag[wave][tl][te] = (_Float16)fmaxf(h1 * s1 + be1v, 0.0f);
    }
  }
  __syncthreads();

  // ---- phase 3: h2 = relu(bn(h1)) @ w_w2 + b2 (K padded 16->32) ----
  v16h af2 = *(const v16h*)&h1frag[wave][lane][0];
  v16h bf2 = *(const v16h*)&w2frag[lane][0];
  v8f acc2 = {};
  acc2 = __builtin_amdgcn_wmma_f32_16x16x32_f16(false, af2, false, bf2, (short)0, acc2, false, false);

  const int n2 = p0 + wave;
  const float b2v = wb2[nn];
#pragma unroll
  for (int i = 0; i < 8; ++i) {
    int m = mh * 8 + i;   // m = ns
    logits[(((size_t)b * NN_ + n2) * NSK + m) * CSK + nn] = acc2[i] + b2v;
  }
}

// ---------------------------------------------------------------------------
// Kernel 4: softmax stats over the N axis (axis=1): per (b,ns,cs) compute
// max and sum(exp) over 8192 points. One WG per (b,ns).
// ---------------------------------------------------------------------------
__global__ __launch_bounds__(256) void softstat_kernel(
    const float* __restrict__ logits, float* __restrict__ stats) {
  __shared__ float red[16][16];
  __shared__ float bcast[16];
  const int b = blockIdx.y, ns = blockIdx.x;
  const int cs = threadIdx.x & 15, g = threadIdx.x >> 4;
  const float* base = logits + (size_t)b * NN_ * NSK * CSK + (size_t)ns * CSK;

  float mx = -3.4e38f;
  for (int n = g; n < NN_; n += 16)
    mx = fmaxf(mx, base[(size_t)n * (NSK * CSK) + cs]);
  red[g][cs] = mx;
  __syncthreads();
  if (threadIdx.x < 16) {
    float m = red[0][threadIdx.x];
#pragma unroll
    for (int i = 1; i < 16; ++i) m = fmaxf(m, red[i][threadIdx.x]);
    bcast[threadIdx.x] = m;
  }
  __syncthreads();
  mx = bcast[cs];

  float sm = 0.0f;
  for (int n = g; n < NN_; n += 16)
    sm += __expf(base[(size_t)n * (NSK * CSK) + cs] - mx);
  __syncthreads();
  red[g][cs] = sm;
  __syncthreads();
  if (threadIdx.x < 16) {
    float s = 0.0f;
#pragma unroll
    for (int i = 0; i < 16; ++i) s += red[i][threadIdx.x];
    size_t o = (((size_t)b * NSK + ns) * CSK + threadIdx.x) * 2;
    stats[o + 0] = bcast[threadIdx.x];
    stats[o + 1] = s;
  }
}

// ---------------------------------------------------------------------------
// Kernel 5: output: out = px + sum_ns attn * (v_gathered + pos_enc), with
// attn = exp(logit - max)/sum broadcast over SH share-groups (cs = c % 16).
// One WG (128 threads = channels) per point.
// ---------------------------------------------------------------------------
__global__ __launch_bounds__(128) void out_kernel(
    const float* __restrict__ px, const float* __restrict__ xyz,
    const int* __restrict__ knn, const float* __restrict__ vb,
    const float* __restrict__ logits, const float* __restrict__ stats,
    const float* __restrict__ pw1, const float* __restrict__ pb1,
    const float* __restrict__ pg1, const float* __restrict__ pbe1,
    const float* __restrict__ pw2, const float* __restrict__ pb2,
    float* __restrict__ out) {
  __shared__ int   idxs[16];
  __shared__ float p1s[16][3];
  __shared__ float sts[16][16][2];

  const int b = blockIdx.y, n = blockIdx.x, c = threadIdx.x;
  const float inv = rsqrtf(1.0f + 1e-5f);

#pragma unroll
  for (int e = 0; e < 4; ++e) {           // stage stats (512 floats)
    int i = e * 128 + c;
    ((float*)sts)[i] = stats[(size_t)b * 512 + i];
  }
  if (c < 16) {                           // per-neighbor pos-enc hidden vec
    const float* xb = xyz + (size_t)b * NN_ * 3;
    int idx = knn[((size_t)b * NN_ + n) * NSK + c];
    idxs[c] = idx;
    float dx = xb[idx * 3 + 0] - xb[n * 3 + 0];
    float dy = xb[idx * 3 + 1] - xb[n * 3 + 1];
    float dz = xb[idx * 3 + 2] - xb[n * 3 + 2];
#pragma unroll
    for (int j = 0; j < 3; ++j) {
      float t = dx * pw1[0 * 3 + j] + dy * pw1[1 * 3 + j] + dz * pw1[2 * 3 + j] + pb1[j];
      p1s[c][j] = fmaxf(t * (pg1[j] * inv) + pbe1[j], 0.0f);
    }
  }
  __syncthreads();

  const int cs = c & 15;
  const float pb2c = pb2[c];
  const float w20 = pw2[c], w21 = pw2[CC + c], w22 = pw2[2 * CC + c];
  float acc = 0.0f;
#pragma unroll
  for (int ns = 0; ns < 16; ++ns) {
    float lw = logits[(((size_t)b * NN_ + n) * NSK + ns) * CSK + cs];
    float a  = __expf(lw - sts[ns][cs][0]) / sts[ns][cs][1];
    float pe = p1s[ns][0] * w20 + p1s[ns][1] * w21 + p1s[ns][2] * w22 + pb2c;
    float vv = vb[((size_t)b * NN_ + idxs[ns]) * CC + c];
    acc += a * (vv + pe);
  }
  size_t o = ((size_t)b * NN_ + n) * CC + c;
  out[o] = px[o] + acc;
}

// ---------------------------------------------------------------------------
extern "C" void kernel_launch(void* const* d_in, const int* in_sizes, int n_in,
                              void* d_out, int out_size, void* d_ws, size_t ws_size,
                              hipStream_t stream) {
  (void)in_sizes; (void)n_in; (void)out_size; (void)ws_size;
  const float* px   = (const float*)d_in[0];
  const float* xyz  = (const float*)d_in[1];
  const float* Wq   = (const float*)d_in[2];  const float* bq   = (const float*)d_in[3];
  const float* Wk   = (const float*)d_in[4];  const float* bk   = (const float*)d_in[5];
  const float* Wv   = (const float*)d_in[6];  const float* bv   = (const float*)d_in[7];
  const float* pw1  = (const float*)d_in[8];  const float* pb1  = (const float*)d_in[9];
  const float* pg1  = (const float*)d_in[10]; const float* pbe1 = (const float*)d_in[11];
  const float* pw2  = (const float*)d_in[12]; const float* pb2  = (const float*)d_in[13];
  const float* wg0  = (const float*)d_in[14]; const float* wbe0 = (const float*)d_in[15];
  const float* ww1  = (const float*)d_in[16]; const float* wb1  = (const float*)d_in[17];
  const float* wg1  = (const float*)d_in[18]; const float* wbe1 = (const float*)d_in[19];
  const float* ww2  = (const float*)d_in[20]; const float* wb2  = (const float*)d_in[21];
  float* out = (float*)d_out;

  const size_t BN = (size_t)BB * NN_;
  float* qb     = (float*)d_ws;                     // BN*128
  float* kb     = qb + BN * CC;                     // BN*128
  float* vb     = kb + BN * CC;                     // BN*128
  int*   knn    = (int*)(vb + BN * CC);             // BN*16
  float* logits = (float*)(knn + BN * NSK);         // BN*16*16
  float* stats  = logits + BN * NSK * CSK;          // B*16*16*2

  knn_kernel   <<<dim3(NN_ / 256, BB),    256, 0, stream>>>(xyz, knn);
  proj_kernel  <<<dim3(BN / 64, 1, 3),    256, 0, stream>>>(px, Wq, bq, Wk, bk, Wv, bv, qb, kb, vb);
  attnw_kernel <<<dim3(NN_ / 8, BB),      256, 0, stream>>>(xyz, knn, qb, kb,
                                                            pw1, pb1, pg1, pbe1, pw2, pb2,
                                                            wg0, wbe0, ww1, wb1, wg1, wbe1, ww2, wb2,
                                                            logits);
  softstat_kernel<<<dim3(NSK, BB),        256, 0, stream>>>(logits, stats);
  out_kernel   <<<dim3(NN_, BB),          128, 0, stream>>>(px, xyz, knn, vb, logits, stats,
                                                            pw1, pb1, pg1, pbe1, pw2, pb2, out);
}
